// BaseDecoder_60524679135703
// MI455X (gfx1250) — matrix operand order repaired and checked by reading, not compiled, not run on hardware
//
#include <hip/hip_runtime.h>

typedef __attribute__((ext_vector_type(16))) __bf16 v16bf;
typedef __attribute__((ext_vector_type(8)))  float  v8f;
typedef __attribute__((ext_vector_type(4)))  unsigned int v4u;

#define MD   4
#define DSP  9          // 2*MD+1 displacements per axis
#define BN   4
#define CCH  128
#define HH   128
#define WW   256
#define TILE 16
#define ROWP 136        // padded LDS row length in bf16 elems (272 B, conflict-free, 16B-aligned)
#define T2W  24         // staged t2 width per row: x0-4 .. x0+19
#define NT   288        // 9 waves of 32

__device__ __forceinline__ unsigned short f2bf(float f) {
    unsigned int u = __float_as_uint(f);
    u += 0x7FFFu + ((u >> 16) & 1u);       // round-to-nearest-even
    return (unsigned short)(u >> 16);
}

union Frag { v4u u2[2]; v16bf v; };

__global__ __launch_bounds__(NT) void corr81_wmma_kernel(
    const float* __restrict__ t1, const float* __restrict__ t2,
    float* __restrict__ out)
{
    __shared__ __align__(16) unsigned short t1S[TILE * ROWP];       // [i][c]
    __shared__ __align__(16) unsigned short t2S[DSP * T2W * ROWP];  // [di][j][c]
    __shared__ __align__(16) float outS[81 * TILE];                 // [p][i]

    const int tid = threadIdx.x;
    const int bx  = blockIdx.x;
    const int xb  = bx & 15;            // WW/16 = 16 tiles per row
    const int y   = (bx >> 4) & 127;    // HH = 128
    const int bb  = bx >> 11;           // / (16*128)
    const int x0  = xb * TILE;

    // ---- stage t1 tile: 16 pixels x 128 channels (f32 -> bf16) ----
    for (int idx = tid; idx < TILE * CCH; idx += NT) {
        int i = idx & 15;
        int c = idx >> 4;
        float v = t1[((bb * CCH + c) * HH + y) * WW + x0 + i];
        t1S[i * ROWP + c] = f2bf(v);
    }
    // ---- stage t2 window: 9 rows x 24 x x 128 c, zero-padded OOB ----
    for (int idx = tid; idx < DSP * T2W * CCH; idx += NT) {
        int j  = idx % T2W;
        int c  = (idx / T2W) & (CCH - 1);
        int di = idx / (T2W * CCH);
        int yy = y + di - MD;
        int xx = x0 - MD + j;
        float v = 0.0f;
        if (yy >= 0 && yy < HH && xx >= 0 && xx < WW)
            v = t2[((bb * CCH + c) * HH + yy) * WW + xx];
        t2S[(di * T2W + j) * ROWP + c] = f2bf(v);
    }
    __syncthreads();

    const int lane = tid & 31;
    const int wave = tid >> 5;      // = di index 0..8
    const int hf   = lane >> 4;     // lane half
    const int mn   = lane & 15;     // M (for A) / N (for B)

    v8f acc0 = {};                  // tile0: x2 = x0-4+N
    v8f acc1 = {};                  // tile1: x2 = x0+4+N

    const unsigned short* t1row = &t1S[mn * ROWP];
    const unsigned short* t2b0  = &t2S[(wave * T2W +     mn) * ROWP];
    const unsigned short* t2b1  = &t2S[(wave * T2W + 8 + mn) * ROWP];

    #pragma unroll
    for (int kc = 0; kc < 4; ++kc) {
        const int kbase = kc * 32;
        Frag fa, fb0, fb1;
        // A (16x32 bf16): VGPR0-7 hold K = kbase + hf*16 + {0..15}
        fa.u2[0] = *(const v4u*)(t1row + kbase + hf * 16);
        fa.u2[1] = *(const v4u*)(t1row + kbase + hf * 16 + 8);
        // B (32x16 bf16): VGPR0-7 hold K = kbase + hf*16 + {0..15}
        fb0.u2[0] = *(const v4u*)(t2b0 + kbase + hf * 16);
        fb0.u2[1] = *(const v4u*)(t2b0 + kbase + hf * 16 + 8);
        fb1.u2[0] = *(const v4u*)(t2b1 + kbase + hf * 16);
        fb1.u2[1] = *(const v4u*)(t2b1 + kbase + hf * 16 + 8);

        acc0 = __builtin_amdgcn_wmma_f32_16x16x32_bf16(
                   false, fa.v, false, fb0.v, (short)0, acc0, false, false);
        acc1 = __builtin_amdgcn_wmma_f32_16x16x32_bf16(
                   false, fa.v, false, fb1.v, (short)0, acc1, false, false);
    }

    // ---- extract the |x2 - x1| <= 4 band into LDS (exact partition, race-free) ----
    #pragma unroll
    for (int r = 0; r < 8; ++r) {
        const int M = r + 8 * hf;   // pixel index within tile
        const int N = mn;           // column within 16-wide B tile
        // tile0: dj = N - 4 - M  (owns M+dj <= 11, i.e. whenever n0 in range)
        int dj0 = N - 4 - M;
        if (dj0 >= -MD && dj0 <= MD)
            outS[(wave * DSP + dj0 + MD) * TILE + M] = acc0[r];
        // tile1: dj = N + 4 - M  (owns M+dj >= 12 <=> N >= 8)
        int dj1 = N + 4 - M;
        if (N >= 8 && dj1 >= -MD && dj1 <= MD)
            outS[(wave * DSP + dj1 + MD) * TILE + M] = acc1[r];
    }
    __syncthreads();

    // ---- coalesced store: 16 consecutive x per displacement plane ----
    for (int idx = tid; idx < 81 * TILE; idx += NT) {
        int p = idx >> 4;
        int i = idx & 15;
        out[((bb * 81 + p) * HH + y) * WW + x0 + i] = outS[idx];
    }
}

extern "C" void kernel_launch(void* const* d_in, const int* in_sizes, int n_in,
                              void* d_out, int out_size, void* d_ws, size_t ws_size,
                              hipStream_t stream) {
    (void)in_sizes; (void)n_in; (void)d_ws; (void)ws_size; (void)out_size;
    const float* t1 = (const float*)d_in[0];
    const float* t2 = (const float*)d_in[1];
    float* out = (float*)d_out;
    const int blocks = BN * HH * (WW / TILE);   // 4*128*16 = 8192
    corr81_wmma_kernel<<<blocks, NT, 0, stream>>>(t1, t2, out);
}
